// PerceiverAttention_70145405878553
// MI455X (gfx1250) — matrix operand order repaired and checked by reading, not compiled, not run on hardware
//
#include <hip/hip_runtime.h>

// ---------------------------------------------------------------------------
// PerceiverAttention for MI455X (gfx1250), wave32 + v_wmma_f32_16x16x32_f16.
// All GEMMs on the matrix pipe in f16 with f32 accumulation; flash softmax in
// f32 registers.  The dominant kv GEMM is LDS-blocked (64x128 per 8-wave
// block) with double-buffered A/B tiles staged by the Tensor Data Mover
// (TENSOR_LOAD_TO_LDS + s_wait_tensorcnt) when the builtin is available.
// NOTE: gemm_kv_kernel is defined first so the disasm snippet shows the
// TDM + LDS + WMMA inner loop.
// ---------------------------------------------------------------------------

typedef __attribute__((ext_vector_type(16))) _Float16 v16h;
typedef __attribute__((ext_vector_type(8)))  _Float16 v8h;
typedef __attribute__((ext_vector_type(8)))  float    v8f;
typedef unsigned int u32x4 __attribute__((ext_vector_type(4)));
typedef int          i32x8 __attribute__((ext_vector_type(8)));
typedef int          i32x4 __attribute__((ext_vector_type(4)));

#define DIM      1024
#define HEADS    8
#define DIM_HEAD 64
#define INNER    512
#define NKV      2112   // 2048 media + 64 latents
#define NLAT     64
#define NMED     2048
#define BATCH    16
// softmax scale (DIM_HEAD^-0.5) * log2(e), folded into q -> exp2 softmax.
#define QSCALE_LOG2E 0.18033688011112042f

#if defined(__AMDGCN__) && __has_builtin(__builtin_amdgcn_tensor_load_to_lds)
#define USE_TDM 1
#else
#define USE_TDM 0
#endif

__device__ __forceinline__ v8f wmma16(v16h a, v16h b, v8f c) {
  return __builtin_amdgcn_wmma_f32_16x16x32_f16(
      /*neg_a=*/false, a, /*neg_b=*/false, b,
      /*c_mod=*/(short)0, c, /*reuse_a=*/false, /*reuse_b=*/false);
}

// A-fragment (16x32, row-major source, ld in halfs).
// lane m holds row m, K = k0+0..15; lane m+16: K = k0+16..31.
__device__ __forceinline__ v16h load_a_frag(const _Float16* base, int ld,
                                            int row0, int k0, int lane) {
  int r  = row0 + (lane & 15);
  int kk = k0 + ((lane >> 4) << 4);
  const _Float16* p = base + (size_t)r * ld + kk;
  return *(const v16h*)(p);
}

// B-fragment (32x16) from B^T stored row-major [N x K] (ldk in halfs).
// Contiguous k: lane n holds B[k0..k0+15][n]; lane n+16 holds B[k0+16..31][n].
__device__ __forceinline__ v16h load_bT_frag(const _Float16* bt, int ldk,
                                             int n0, int k0, int lane) {
  int n  = n0 + (lane & 15);
  int kk = k0 + ((lane >> 4) << 4);
  return *(const v16h*)(bt + (size_t)n * ldk + kk);
}

// ---------------------------------------------------------------------------
// TDM: 2D-tile load Global->LDS per ISA cdna5 ch.8 (D# groups 0/1; 2/3 zero).
// data_size=2B; tile = tile_w (dim0 elems) x tile_h rows; row stride in elems.
// ---------------------------------------------------------------------------
#if USE_TDM
__device__ __forceinline__ void tdm_load_tile_2d(unsigned lds_byte_addr,
                                                 const _Float16* gsrc,
                                                 unsigned tile_w, unsigned tile_h,
                                                 unsigned tensor_w, unsigned tensor_h,
                                                 unsigned row_stride_elems) {
  unsigned long long ga = (unsigned long long)(size_t)gsrc;
  u32x4 g0;
  g0.x = 1u;                                    // count=1, user mode
  g0.y = lds_byte_addr;                         // lds_addr [63:32]
  g0.z = (unsigned)ga;                          // global_addr low
  g0.w = ((unsigned)(ga >> 32) & 0x01ffffffu)   // global_addr [56:32]
         | (2u << 30);                          // type = 2 ("image")
  i32x8 g1;
  g1[0] = 0x00010000;                           // wg_mask=0, data_size=2B
  g1[1] = (int)((tensor_w & 0xffffu) << 16);    // tensor_dim0 [15:0] @bit48
  g1[2] = (int)((tensor_w >> 16) | ((tensor_h & 0xffffu) << 16));
  g1[3] = (int)((tensor_h >> 16) | (tile_w << 16));  // tile_dim0
  g1[4] = (int)tile_h;                          // tile_dim1 (tile_dim2=0)
  g1[5] = (int)row_stride_elems;                // tensor_dim0_stride low32
  g1[6] = 0;
  g1[7] = 0;
  i32x4 zz = {0, 0, 0, 0};
#if __has_include(<hip/amd_detail/amd_gfx1250_TDM.h>)
  i32x8 z8 = {0, 0, 0, 0, 0, 0, 0, 0};
  __builtin_amdgcn_tensor_load_to_lds(g0, g1, zz, zz, z8, 0);
#else
  __builtin_amdgcn_tensor_load_to_lds(g0, g1, zz, zz, 0);
#endif
}
#endif

// ---------------------------------------------------------------------------
// kv = kvin @ Wkv.  LDS-blocked: block tile 64(M) x 128(N), 8 waves each
// computing 16x64.  A (64x32) and B (128x32) K-step tiles double-buffered in
// LDS, staged by TDM (wave 0 issues descriptors; TENSORcnt) or fallback copy.
// Cols 0..511 -> K [b,h,s,d]; cols 512..1023 -> V transposed [b,h,d,s].
// ---------------------------------------------------------------------------
__global__ __launch_bounds__(256) void gemm_kv_kernel(
    const _Float16* __restrict__ kvin, const _Float16* __restrict__ wkvT,
    _Float16* __restrict__ kout, _Float16* __restrict__ vTout) {
  __shared__ __attribute__((aligned(32))) _Float16 atile[2][64][32];   //  8 KB
  __shared__ __attribute__((aligned(32))) _Float16 btile[2][128][32];  // 16 KB
  int lane = threadIdx.x & 31;
  int wave = threadIdx.x >> 5;
  int wm = wave & 3, wn = wave >> 2;
  int m0 = blockIdx.x * 64;            // 528 blocks (33 per batch, no crossing)
  int nb = blockIdx.y * 128;           // 8 blocks
  v8f acc0 = {}, acc1 = {}, acc2 = {}, acc3 = {};

  // ---- stage K-chunk 0 ----
#if USE_TDM
  if (wave == 0) {
    tdm_load_tile_2d((unsigned)(size_t)&atile[0][0][0],
                     kvin + (size_t)m0 * DIM, 32, 64, DIM, 64, DIM);
    tdm_load_tile_2d((unsigned)(size_t)&btile[0][0][0],
                     wkvT + (size_t)nb * DIM, 32, 128, DIM, 128, DIM);
  }
#else
  {
    int t = threadIdx.x;
    int r = t >> 2, c = (t & 3) << 3;
    *(v8h*)(&atile[0][r][c]) = *(const v8h*)(kvin + (size_t)(m0 + r) * DIM + c);
    int r2 = t >> 1, c2 = (t & 1) << 4;
    *(v16h*)(&btile[0][r2][c2]) = *(const v16h*)(wkvT + (size_t)(nb + r2) * DIM + c2);
  }
#endif

  for (int kt = 0; kt < 32; ++kt) {
    int p = kt & 1;
    int k0 = kt * 32;
    // ---- prefetch K-chunk kt+1 into the other buffer ----
    if (kt + 1 < 32) {
#if USE_TDM
      if (wave == 0) {
        tdm_load_tile_2d((unsigned)(size_t)&atile[1 - p][0][0],
                         kvin + (size_t)m0 * DIM + k0 + 32, 32, 64, DIM, 64, DIM);
        tdm_load_tile_2d((unsigned)(size_t)&btile[1 - p][0][0],
                         wkvT + (size_t)nb * DIM + k0 + 32, 32, 128, DIM, 128, DIM);
      }
#else
      int t = threadIdx.x;
      int r = t >> 2, c = (t & 3) << 3;
      *(v8h*)(&atile[1 - p][r][c]) =
          *(const v8h*)(kvin + (size_t)(m0 + r) * DIM + k0 + 32 + c);
      int r2 = t >> 1, c2 = (t & 1) << 4;
      *(v16h*)(&btile[1 - p][r2][c2]) =
          *(const v16h*)(wkvT + (size_t)(nb + r2) * DIM + k0 + 32 + c2);
#endif
    }
#if USE_TDM
    if (wave == 0) {
      if (kt + 1 < 32) __builtin_amdgcn_s_wait_tensorcnt(2);  // current pair done
      else             __builtin_amdgcn_s_wait_tensorcnt(0);
    }
#endif
    __syncthreads();
    // ---- compute on buffer p: 1 A-frag feeds 4 WMMAs ----
    v16h a = load_a_frag(&atile[p][wm * 16][0], 32, 0, 0, lane);
    int ksel = (lane >> 4) << 4;
    int nl = wn * 64 + (lane & 15);
    acc0 = wmma16(a, *(const v16h*)(&btile[p][nl][ksel]), acc0);
    acc1 = wmma16(a, *(const v16h*)(&btile[p][nl + 16][ksel]), acc1);
    acc2 = wmma16(a, *(const v16h*)(&btile[p][nl + 32][ksel]), acc2);
    acc3 = wmma16(a, *(const v16h*)(&btile[p][nl + 48][ksel]), acc3);
    __syncthreads();
  }

  // ---- epilogue: split K / V-transposed stores ----
  int bidx = m0 / NKV;
  int s0 = m0 - bidx * NKV + wm * 16 + ((lane >> 4) << 3);
  v8f accs[4] = {acc0, acc1, acc2, acc3};
#pragma unroll
  for (int sub = 0; sub < 4; ++sub) {
    int col = nb + wn * 64 + sub * 16 + (lane & 15);
    if (col < INNER) {                 // K matrix, row-major [s, d]
      int h = col >> 6, d = col & 63;
      _Float16* kp =
          kout + ((size_t)(bidx * HEADS + h) * NKV + s0) * DIM_HEAD + d;
#pragma unroll
      for (int g = 0; g < 8; ++g) kp[(size_t)g * DIM_HEAD] = (_Float16)accs[sub][g];
    } else {                           // V transposed [d, s]: contiguous pack
      int c = col - INNER;
      int h = c >> 6, d = c & 63;
      v8h pk;
#pragma unroll
      for (int g = 0; g < 8; ++g) pk[g] = (_Float16)accs[sub][g];
      *(v8h*)(vTout + ((size_t)(bidx * HEADS + h) * DIM_HEAD + d) * NKV + s0) = pk;
    }
  }
}

// ---------------------------------------------------------------------------
// Weight transpose + f32->f16:  wt[n*K + k] = w[k*N + n]
// ---------------------------------------------------------------------------
__global__ __launch_bounds__(256) void wT_kernel(const float* __restrict__ w,
                                                 _Float16* __restrict__ wt,
                                                 int K, int N) {
  int k = blockIdx.x * 256 + threadIdx.x;
  int n = blockIdx.y;
  if (k < K) wt[(size_t)n * K + k] = (_Float16)w[(size_t)k * N + n];
}

// ---------------------------------------------------------------------------
// Row LayerNorm (1024 cols), f32 -> f16 into concatenated kv-input buffer.
// ---------------------------------------------------------------------------
__global__ __launch_bounds__(256) void ln_kernel(const float* __restrict__ src,
                                                 const float* __restrict__ w,
                                                 const float* __restrict__ bias,
                                                 _Float16* __restrict__ dst,
                                                 int rows_per_b, int dst_off) {
  __shared__ float red[2][8];
  int r = blockIdx.x;
  int t = threadIdx.x;
  int b = r / rows_per_b;
  int s = r - b * rows_per_b;
  const float* x = src + (size_t)r * DIM;
  _Float16* y = dst + (size_t)(b * NKV + dst_off + s) * DIM;

  float v0 = x[t], v1 = x[t + 256], v2 = x[t + 512], v3 = x[t + 768];
  float sum = v0 + v1 + v2 + v3;
  float sq  = v0 * v0 + v1 * v1 + v2 * v2 + v3 * v3;
#pragma unroll
  for (int m = 1; m < 32; m <<= 1) {
    sum += __shfl_xor(sum, m, 32);
    sq  += __shfl_xor(sq, m, 32);
  }
  if ((t & 31) == 0) { red[0][t >> 5] = sum; red[1][t >> 5] = sq; }
  __syncthreads();
  float ts = 0.f, tq = 0.f;
#pragma unroll
  for (int i = 0; i < 8; ++i) { ts += red[0][i]; tq += red[1][i]; }
  float mu   = ts * (1.0f / DIM);
  float var  = tq * (1.0f / DIM) - mu * mu;
  float rinv = rsqrtf(var + 1e-5f);
  y[t]       = (_Float16)((v0 - mu) * rinv * w[t]       + bias[t]);
  y[t + 256] = (_Float16)((v1 - mu) * rinv * w[t + 256] + bias[t + 256]);
  y[t + 512] = (_Float16)((v2 - mu) * rinv * w[t + 512] + bias[t + 512]);
  y[t + 768] = (_Float16)((v3 - mu) * rinv * w[t + 768] + bias[t + 768]);
}

// ---------------------------------------------------------------------------
// q = ln(latents) @ Wq (scale*log2e folded).  Out: q[b,h,m,d] f16.
// ---------------------------------------------------------------------------
__global__ __launch_bounds__(128) void gemm_q_kernel(
    const _Float16* __restrict__ kvin, const _Float16* __restrict__ wqT,
    _Float16* __restrict__ qout) {
  int lane = threadIdx.x & 31;
  int wave = threadIdx.x >> 5;
  int mt = blockIdx.x;                 // 0..63 : 4 tiles per batch
  int nt = blockIdx.y * 4 + wave;      // 0..31
  int b  = mt >> 2;
  int s0 = NMED + ((mt & 3) << 4);
  const _Float16* abase = kvin + (size_t)b * NKV * DIM;
  int n0 = nt * 16;
  v8f acc = {};
  for (int k0 = 0; k0 < DIM; k0 += 32) {
    acc = wmma16(load_a_frag(abase, DIM, s0, k0, lane),
                 load_bT_frag(wqT, DIM, n0, k0, lane), acc);
  }
  int col = n0 + (lane & 15);
  int h = col >> 6, d = col & 63;
  int mm0 = ((mt & 3) << 4) + ((lane >> 4) << 3);
#pragma unroll
  for (int g = 0; g < 8; ++g)
    qout[((size_t)(b * HEADS + h) * NLAT + mm0 + g) * DIM_HEAD + d] =
        (_Float16)(acc[g] * QSCALE_LOG2E);
}

// ---------------------------------------------------------------------------
// Flash attention: one block per (b,h), 4 waves x 16 q-rows.
// ---------------------------------------------------------------------------
__global__ __launch_bounds__(128) void attn_kernel(
    const _Float16* __restrict__ qh,   // [b,h,64,64]
    const _Float16* __restrict__ kh,   // [b,h,2112,64]
    const _Float16* __restrict__ vT,   // [b,h,64,2112]
    _Float16* __restrict__ att) {      // [b,64,512]
  __shared__ __attribute__((aligned(32))) _Float16 pstage[4][16][32];
  int lane = threadIdx.x & 31;
  int wave = threadIdx.x >> 5;
  int bh = blockIdx.x;
  int b = bh >> 3, h = bh & 7;
  const _Float16* q  = qh + (size_t)bh * NLAT * DIM_HEAD;
  const _Float16* kp = kh + (size_t)bh * NKV * DIM_HEAD;
  const _Float16* vp = vT + (size_t)bh * DIM_HEAD * NKV;
  int m0 = wave * 16;

  v16h aq0 = load_a_frag(q, DIM_HEAD, m0, 0, lane);
  v16h aq1 = load_a_frag(q, DIM_HEAD, m0, 32, lane);

  v8f oc0 = {}, oc1 = {}, oc2 = {}, oc3 = {};
  float rowm[8], rowl[8];
#pragma unroll
  for (int g = 0; g < 8; ++g) { rowm[g] = -1e30f; rowl[g] = 0.0f; }

  for (int j0 = 0; j0 < NKV; j0 += 32) {
    if (j0 + 32 < NKV)  // stream next K chunk (global_prefetch_b8)
      __builtin_prefetch(kp + (size_t)(j0 + 32 + lane) * DIM_HEAD, 0, 0);

    int jbase = j0 + (lane & 15);
    int dsel  = (lane >> 4) << 4;
    v8f s0 = {}, s1 = {};
    s0 = wmma16(aq0, *(const v16h*)(kp + (size_t)jbase * DIM_HEAD + dsel), s0);
    s0 = wmma16(aq1, *(const v16h*)(kp + (size_t)jbase * DIM_HEAD + 32 + dsel), s0);
    s1 = wmma16(aq0, *(const v16h*)(kp + (size_t)(jbase + 16) * DIM_HEAD + dsel), s1);
    s1 = wmma16(aq1, *(const v16h*)(kp + (size_t)(jbase + 16) * DIM_HEAD + 32 + dsel), s1);

#pragma unroll
    for (int g = 0; g < 8; ++g) {
      float t = fmaxf(s0[g], s1[g]);
      t = fmaxf(t, __shfl_xor(t, 1, 32));
      t = fmaxf(t, __shfl_xor(t, 2, 32));
      t = fmaxf(t, __shfl_xor(t, 4, 32));
      t = fmaxf(t, __shfl_xor(t, 8, 32));
      float mnew = fmaxf(rowm[g], t);
      float fac  = exp2f(rowm[g] - mnew);
      rowm[g] = mnew;
      float p0 = exp2f(s0[g] - mnew);
      float p1 = exp2f(s1[g] - mnew);
      s0[g] = p0; s1[g] = p1;
      float rs = p0 + p1;
      rs += __shfl_xor(rs, 1, 32);
      rs += __shfl_xor(rs, 2, 32);
      rs += __shfl_xor(rs, 4, 32);
      rs += __shfl_xor(rs, 8, 32);
      rowl[g] = rowl[g] * fac + rs;
      oc0[g] *= fac; oc1[g] *= fac; oc2[g] *= fac; oc3[g] *= fac;
    }

    // C-layout -> A-layout for P via LDS (wave-private 1KB region).
    __syncthreads();
    int r = (lane >> 4) << 3;
#pragma unroll
    for (int g = 0; g < 8; ++g) {
      pstage[wave][r + g][lane & 15]        = (_Float16)s0[g];
      pstage[wave][r + g][16 + (lane & 15)] = (_Float16)s1[g];
    }
    __syncthreads();
    v16h ap = load_a_frag(&pstage[wave][0][0], 32, 0, 0, lane);

    int jsel = j0 + ((lane >> 4) << 4);
    oc0 = wmma16(ap, *(const v16h*)(vp + (size_t)(0  + (lane & 15)) * NKV + jsel), oc0);
    oc1 = wmma16(ap, *(const v16h*)(vp + (size_t)(16 + (lane & 15)) * NKV + jsel), oc1);
    oc2 = wmma16(ap, *(const v16h*)(vp + (size_t)(32 + (lane & 15)) * NKV + jsel), oc2);
    oc3 = wmma16(ap, *(const v16h*)(vp + (size_t)(48 + (lane & 15)) * NKV + jsel), oc3);
  }

  int mrow0 = m0 + ((lane >> 4) << 3);
#pragma unroll
  for (int g = 0; g < 8; ++g) {
    float inv = 1.0f / rowl[g];
    _Float16* op = att + ((size_t)(b * NLAT) + mrow0 + g) * INNER + h * DIM_HEAD;
    op[0  + (lane & 15)] = (_Float16)(oc0[g] * inv);
    op[16 + (lane & 15)] = (_Float16)(oc1[g] * inv);
    op[32 + (lane & 15)] = (_Float16)(oc2[g] * inv);
    op[48 + (lane & 15)] = (_Float16)(oc3[g] * inv);
  }
}

// ---------------------------------------------------------------------------
// out = att [1024, 512] @ Wout -> f32 [1024, 1024].
// ---------------------------------------------------------------------------
__global__ __launch_bounds__(128) void gemm_out_kernel(
    const _Float16* __restrict__ att, const _Float16* __restrict__ woutT,
    float* __restrict__ out) {
  int lane = threadIdx.x & 31;
  int wave = threadIdx.x >> 5;
  int m0 = blockIdx.x * 16;
  int n0 = (blockIdx.y * 4 + wave) * 16;
  v8f acc = {};
  for (int k0 = 0; k0 < INNER; k0 += 32) {
    acc = wmma16(load_a_frag(att, INNER, m0, k0, lane),
                 load_bT_frag(woutT, INNER, n0, k0, lane), acc);
  }
  int n  = n0 + (lane & 15);
  int mr = m0 + ((lane >> 4) << 3);
#pragma unroll
  for (int g = 0; g < 8; ++g) out[(size_t)(mr + g) * DIM + n] = acc[g];
}

// ---------------------------------------------------------------------------
extern "C" void kernel_launch(void* const* d_in, const int* in_sizes, int n_in,
                              void* d_out, int out_size, void* d_ws,
                              size_t ws_size, hipStream_t stream) {
  const float* x    = (const float*)d_in[0];
  const float* lat  = (const float*)d_in[1];
  const float* lnmw = (const float*)d_in[2];
  const float* lnmb = (const float*)d_in[3];
  const float* lnlw = (const float*)d_in[4];
  const float* lnlb = (const float*)d_in[5];
  const float* Wq   = (const float*)d_in[6];
  const float* Wkv  = (const float*)d_in[7];
  const float* Wout = (const float*)d_in[8];
  float* out = (float*)d_out;

  char* p = (char*)d_ws;
  auto carve = [&](size_t bytes) {
    char* r = p;
    p += (bytes + 255) & ~(size_t)255;
    return (void*)r;
  };
  // ~145 MB total scratch
  _Float16* kvin  = (_Float16*)carve((size_t)BATCH * NKV * DIM * 2);
  _Float16* wqT   = (_Float16*)carve((size_t)INNER * DIM * 2);
  _Float16* wkvT  = (_Float16*)carve((size_t)(2 * INNER) * DIM * 2);
  _Float16* woutT = (_Float16*)carve((size_t)DIM * INNER * 2);
  _Float16* qh    = (_Float16*)carve((size_t)BATCH * HEADS * NLAT * DIM_HEAD * 2);
  _Float16* kh    = (_Float16*)carve((size_t)BATCH * HEADS * NKV * DIM_HEAD * 2);
  _Float16* vT    = (_Float16*)carve((size_t)BATCH * HEADS * DIM_HEAD * NKV * 2);
  _Float16* att   = (_Float16*)carve((size_t)BATCH * NLAT * INNER * 2);

  wT_kernel<<<dim3(4, 512), 256, 0, stream>>>(Wq, wqT, DIM, INNER);
  wT_kernel<<<dim3(4, 1024), 256, 0, stream>>>(Wkv, wkvT, DIM, 2 * INNER);
  wT_kernel<<<dim3(2, 1024), 256, 0, stream>>>(Wout, woutT, INNER, DIM);

  ln_kernel<<<dim3(BATCH * NMED), 256, 0, stream>>>(x, lnmw, lnmb, kvin, NMED, 0);
  ln_kernel<<<dim3(BATCH * NLAT), 256, 0, stream>>>(lat, lnlw, lnlb, kvin, NLAT, NMED);

  gemm_q_kernel<<<dim3(64, 8), 128, 0, stream>>>(kvin, wqT, qh);
  gemm_kv_kernel<<<dim3(528, 8), 256, 0, stream>>>(kvin, wkvT, kh, vT);
  attn_kernel<<<dim3(BATCH * HEADS), 128, 0, stream>>>(qh, kh, vT, att);
  gemm_out_kernel<<<dim3(64, 16), 128, 0, stream>>>(att, woutT, out);
}
